// CMFlow_T_81260781240445
// MI455X (gfx1250) — matrix operand order repaired and checked by reading, not compiled, not run on hardware
//
#include <hip/hip_runtime.h>
#include <hip/hip_bf16.h>
#include <math.h>

// ============================================================================
// CMFlow forward for MI455X (gfx1250, wave32).
// Dense math on v_wmma_f32_16x16x32_f16: one wave computes a 16x64 output
// tile (A fragment reused across 4 WMMAs), K-step 32, f32 accumulate.
// Weights are pre-transposed+converted to f16 [N][Kpad] with zero padding to
// a multiple of 32 -> each B fragment is two 16B-aligned 16B reads
// (2x b128) and needs no K guards. M/N edges by address clamping (garbage
// only lands in unstored outputs); A K-tail in a separate guarded block.
// Gather/concat stages are fused into the WMMA A-operand ("virtual A") to
// avoid materializing the 135MB mse2 inputs. Weights (~10MB) L2-resident.
// ============================================================================

typedef __attribute__((ext_vector_type(16))) _Float16 v16h;
typedef __attribute__((ext_vector_type(8)))  float    v8f;

#define BB 2
#define NNP 512   // points per cloud

// ---------------------------------------------------------------------------
// A-operand loaders. frag16(row,kstart) fills the WMMA A fragment
// (elems 0..7 -> kstart..kstart+7, elems 8..15 -> kstart+16..kstart+23);
// one(row,k) is the scalar fallback for the K tail. row pre-clamped.
// ---------------------------------------------------------------------------
struct DenseA {
  const float* A; int K;
  __device__ float one(int row, int k) const { return A[(size_t)row * K + k]; }
  __device__ v16h frag16(int row, int kstart) const {
    const float4* q = (const float4*)(A + (size_t)row * K + kstart);  // 16B aligned: K%4==0
    float4 a0 = q[0], a1 = q[1], b0 = q[4], b1 = q[5];
    v16h r;
    r[0] = (_Float16)a0.x; r[1] = (_Float16)a0.y; r[2] = (_Float16)a0.z; r[3] = (_Float16)a0.w;
    r[4] = (_Float16)a1.x; r[5] = (_Float16)a1.y; r[6] = (_Float16)a1.z; r[7] = (_Float16)a1.w;
    r[8] = (_Float16)b0.x; r[9] = (_Float16)b0.y; r[10] = (_Float16)b0.z; r[11] = (_Float16)b0.w;
    r[12] = (_Float16)b1.x; r[13] = (_Float16)b1.y; r[14] = (_Float16)b1.z; r[15] = (_Float16)b1.w;
    return r;
  }
};

// point-local-feature concat: [g_xyz(3) | g_feat(Cf)] gathered via ball-query idx
struct GatherPLF {
  const float* xyz; const float* feat; const int* idx;
  int ns; int Cf;
  __device__ float one(int row, int k) const {
    int s  = row % ns;
    int pn = (row / ns) % NNP;
    int b  = row / (ns * NNP);
    int j  = idx[((size_t)b * NNP + pn) * ns + s];
    if (k < 3) return xyz[((size_t)b * NNP + j) * 3 + k] -
                      xyz[((size_t)b * NNP + pn) * 3 + k];
    return feat[((size_t)b * NNP + j) * Cf + (k - 3)];
  }
  __device__ v16h frag16(int row, int kstart) const {
    v16h r;
#pragma unroll
    for (int e = 0; e < 8; ++e)  r[e]     = (_Float16)one(row, kstart + e);
#pragma unroll
    for (int e = 0; e < 8; ++e)  r[8 + e] = (_Float16)one(row, kstart + 16 + e);
    return r;
  }
};

// correlator concat: [f1(n)(C) | f2(j)(C) | p2(j)-p1(n) (3)] via knn idx
struct GatherCorr {
  const float* p1; const float* p2; const float* f1; const float* f2;
  const int* idx; int kk; int C;
  __device__ float one(int row, int k) const {
    int s  = row % kk;
    int pn = (row / kk) % NNP;
    int b  = row / (kk * NNP);
    int j  = idx[((size_t)b * NNP + pn) * kk + s];
    if (k < C)     return f1[((size_t)b * NNP + pn) * C + k];
    if (k < 2 * C) return f2[((size_t)b * NNP + j) * C + (k - C)];
    int c = k - 2 * C;
    return p2[((size_t)b * NNP + j) * 3 + c] - p1[((size_t)b * NNP + pn) * 3 + c];
  }
  __device__ v16h frag16(int row, int kstart) const {
    v16h r;
#pragma unroll
    for (int e = 0; e < 8; ++e)  r[e]     = (_Float16)one(row, kstart + e);
#pragma unroll
    for (int e = 0; e < 8; ++e)  r[8 + e] = (_Float16)one(row, kstart + 16 + e);
    return r;
  }
};

// ---------------------------------------------------------------------------
// W pre-pass: f32 [K][N] row-major -> f16 [N][Kpad] (transposed, zero padded)
// ---------------------------------------------------------------------------
__global__ void convert_w_kernel(const float* __restrict__ W, _Float16* __restrict__ Wt,
                                 int K, int N, int Kpad) {
  int i = blockIdx.x * blockDim.x + threadIdx.x;
  if (i >= N * Kpad) return;
  int n = i / Kpad, k = i % Kpad;
  Wt[i] = (_Float16)((k < K) ? W[(size_t)k * N + n] : 0.0f);
}

// ---------------------------------------------------------------------------
// WMMA GEMM:  C[M,Ncols] = act(A[M,K] @ W[K,Ncols] + bias)
// act: 0=none 1=relu 2=sigmoid.  One wave per 16x64 tile, wave32.
// Fragment packing per CDNA5 ISA 7.12.2:
//   A lane L: row = L&15, elem e -> K = (L&16?8:0) + (e<8 ? e : e+8)
//   B lane L: col = L&15, elem e -> K = (L&16?8:0) + (e<8 ? e : e+8)
//   D lane L: col = L&15, elem v -> row = v + 8*(L>>4)
// ---------------------------------------------------------------------------
template <class ALoad>
__global__ void __launch_bounds__(32)
gemm_wmma(ALoad aload, const _Float16* __restrict__ Wt, const float* __restrict__ bias,
          float* __restrict__ C, int M, int K, int Kpad, int Ncols, int ldC, int act)
{
  const int lane  = threadIdx.x;
  const int row   = blockIdx.x * 16 + (lane & 15);
  const int rowc  = (row < M) ? row : (M - 1);          // clamp: safe, unstored
  const int abase = (lane & 16) ? 8 : 0;
  const int bbase = (lane & 16) ? 8 : 0;
  const int coln  = lane & 15;
  const int colb  = blockIdx.y * 64;

  const _Float16* bp[4];
#pragma unroll
  for (int t = 0; t < 4; ++t) {
    int c  = colb + t * 16 + coln;
    int cc = (c < Ncols) ? c : (Ncols - 1);             // clamp: safe, unstored
    bp[t] = Wt + (size_t)cc * Kpad + bbase;             // 16B aligned stream
  }

  v8f acc[4];
  acc[0] = {}; acc[1] = {}; acc[2] = {}; acc[3] = {};

  int k0 = 0;
  // ---- main loop: full 32-wide K blocks, branch/select free ----
  for (; k0 + 32 <= K; k0 += 32) {
    if (k0 + 64 <= K)
      __builtin_prefetch(bp[0] + k0 + 32, 0, 1);
    v16h af = aload.frag16(rowc, k0 + abase);
    v16h bf[4];
#pragma unroll
    for (int t = 0; t < 4; ++t) { const _Float16* p = bp[t] + k0; for (int e = 0; e < 8; ++e) { bf[t][e] = p[e]; bf[t][8 + e] = p[16 + e]; } }
#pragma unroll
    for (int t = 0; t < 4; ++t)
      acc[t] = __builtin_amdgcn_wmma_f32_16x16x32_f16(false, af, false, bf[t],
                                                      (short)0, acc[t], false, false);
  }
  // ---- K tail: A guarded (clamp + zero-select); B reads zero padding ----
  if (k0 < K) {
    v16h af;
#pragma unroll
    for (int e = 0; e < 16; ++e) {
      int ka  = k0 + abase + ((e < 8) ? e : (e + 8));
      int kac = (ka < K) ? ka : (K - 1);
      float v = aload.one(rowc, kac);
      af[e] = (_Float16)((ka < K) ? v : 0.0f);
    }
    v16h bf[4];
#pragma unroll
    for (int t = 0; t < 4; ++t) { const _Float16* p = bp[t] + k0; for (int e = 0; e < 8; ++e) { bf[t][e] = p[e]; bf[t][8 + e] = p[16 + e]; } }
#pragma unroll
    for (int t = 0; t < 4; ++t)
      acc[t] = __builtin_amdgcn_wmma_f32_16x16x32_f16(false, af, false, bf[t],
                                                      (short)0, acc[t], false, false);
  }
  // ---- store with bias + activation ----
  const int rbase = blockIdx.x * 16 + ((lane & 16) ? 8 : 0);
#pragma unroll
  for (int t = 0; t < 4; ++t) {
    int col = colb + t * 16 + coln;
    if (col < Ncols) {
      float bv = bias ? bias[col] : 0.0f;
#pragma unroll
      for (int v = 0; v < 8; ++v) {
        int r = rbase + v;
        if (r < M) {
          float x = acc[t][v] + bv;
          if (act == 1)      x = fmaxf(x, 0.0f);
          else if (act == 2) x = 1.0f / (1.0f + __expf(-x));
          C[(size_t)r * ldC + col] = x;
        }
      }
    }
  }
}

// ---------------------------------------------------------------------------
// Elementwise / reduction helpers
// ---------------------------------------------------------------------------
__global__ void transpose_kernel(const float* __restrict__ src, float* __restrict__ dst) {
  int i = blockIdx.x * blockDim.x + threadIdx.x;          // src (B,3,N) -> dst (B,N,3)
  if (i >= BB * 3 * NNP) return;
  int n = i % NNP; int c = (i / NNP) % 3; int b = i / (3 * NNP);
  dst[((size_t)b * NNP + n) * 3 + c] = src[i];
}

// reference ball query: first `ns` qualifying indices ascending, padded with
// the first qualifying index (self always qualifies since d=0 <= r^2)
__global__ void ball_query_kernel(const float* __restrict__ p, int* __restrict__ idx,
                                  float r2, int ns) {
  int i = blockIdx.x * blockDim.x + threadIdx.x;
  if (i >= BB * NNP) return;
  int b = i / NNP, n = i % NNP;
  const float* base = p + (size_t)b * NNP * 3;
  float x = base[n * 3], y = base[n * 3 + 1], z = base[n * 3 + 2];
  int* out = idx + (size_t)i * ns;
  int cnt = 0, first = n;
  for (int j = 0; j < NNP; ++j) {
    float dx = base[j * 3] - x, dy = base[j * 3 + 1] - y, dz = base[j * 3 + 2] - z;
    float d = dx * dx + dy * dy + dz * dz;
    if (d <= r2) {
      if (cnt == 0) first = j;
      if (cnt < ns) out[cnt++] = j;
      if (cnt == ns) break;
    }
  }
  for (int s = cnt; s < ns; ++s) out[s] = first;
}

__global__ void knn_kernel(const float* __restrict__ p1, const float* __restrict__ p2,
                           int* __restrict__ idx) {
  int i = blockIdx.x * blockDim.x + threadIdx.x;
  if (i >= BB * NNP) return;
  int b = i / NNP, n = i % NNP;
  const float* q = p1 + ((size_t)b * NNP + n) * 3;
  const float* base = p2 + (size_t)b * NNP * 3;
  float bd[8]; int bi[8];
  for (int s = 0; s < 8; ++s) { bd[s] = 3.4e38f; bi[s] = 0; }
  for (int j = 0; j < NNP; ++j) {
    float dx = base[j * 3] - q[0], dy = base[j * 3 + 1] - q[1], dz = base[j * 3 + 2] - q[2];
    float d = dx * dx + dy * dy + dz * dz;
    if (d < bd[7]) {
      int s = 7;
      while (s > 0 && bd[s - 1] > d) { bd[s] = bd[s - 1]; bi[s] = bi[s - 1]; --s; }
      bd[s] = d; bi[s] = j;
    }
  }
  for (int s = 0; s < 8; ++s) idx[(size_t)i * 8 + s] = bi[s];
}

__global__ void group_max_kernel(const float* __restrict__ X, float* __restrict__ out,
                                 int G, int ns, int C) {
  int i = blockIdx.x * blockDim.x + threadIdx.x;
  if (i >= G * C) return;
  int g = i / C, c = i % C;
  const float* p = X + (size_t)g * ns * C + c;
  float m = p[0];
  for (int s = 1; s < ns; ++s) m = fmaxf(m, p[(size_t)s * C]);
  out[(size_t)g * C + c] = m;
}

// f (B,N,2*Ch): cols [Ch,2Ch) = max over N of cols [0,Ch)
__global__ void colmax_bcast_kernel(float* __restrict__ f, int Ch) {
  int i = blockIdx.x * blockDim.x + threadIdx.x;
  if (i >= BB * Ch) return;
  int b = i / Ch, c = i % Ch;
  float m = -3.4e38f;
  for (int n = 0; n < NNP; ++n) m = fmaxf(m, f[((size_t)b * NNP + n) * (2 * Ch) + c]);
  for (int n = 0; n < NNP; ++n) f[((size_t)b * NNP + n) * (2 * Ch) + Ch + c] = m;
}

__global__ void build_emb_kernel(const float* __restrict__ ft1, const float* __restrict__ f1,
                                 const float* __restrict__ cor, float* __restrict__ emb) {
  int i = blockIdx.x * blockDim.x + threadIdx.x;
  if (i >= BB * NNP * 1027) return;
  int c = i % 1027; int bn = i / 1027;
  float v;
  if (c < 3)        v = ft1[(size_t)bn * 3 + c];
  else if (c < 515) v = f1[(size_t)bn * 512 + (c - 3)];
  else              v = cor[(size_t)bn * 512 + (c - 515)];
  emb[i] = v;
}

__global__ void max_over_n_kernel(const float* __restrict__ x, float* __restrict__ g, int C) {
  int i = blockIdx.x * blockDim.x + threadIdx.x;
  if (i >= BB * C) return;
  int b = i / C, c = i % C;
  float m = -3.4e38f;
  for (int n = 0; n < NNP; ++n) m = fmaxf(m, x[((size_t)b * NNP + n) * C + c]);
  g[i] = m;
}

__global__ void gru_combine_kernel(const float* __restrict__ gi, const float* __restrict__ gh,
                                   const float* __restrict__ h0, float* __restrict__ h) {
  int i = blockIdx.x * blockDim.x + threadIdx.x;
  if (i >= BB * 256) return;
  int b = i / 256, c = i % 256;
  float ir = gi[b * 768 + c],       iz = gi[b * 768 + 256 + c], in = gi[b * 768 + 512 + c];
  float hr = gh[b * 768 + c],       hz = gh[b * 768 + 256 + c], hn = gh[b * 768 + 512 + c];
  float r = 1.0f / (1.0f + __expf(-(ir + hr)));
  float z = 1.0f / (1.0f + __expf(-(iz + hz)));
  float nn = tanhf(in + r * hn);
  h[i] = (1.0f - z) * nn + z * h0[i];
}

__global__ void build_final_kernel(const float* __restrict__ prop, const float* __restrict__ h,
                                   float* __restrict__ fin) {
  int i = blockIdx.x * blockDim.x + threadIdx.x;
  if (i >= BB * NNP * 512) return;
  int c = i % 512; int bn = i / 512; int b = bn / NNP;
  fin[i] = (c < 256) ? prop[(size_t)bn * 256 + c] : h[b * 256 + (c - 256)];
}

__global__ void score_sum_kernel(const float* __restrict__ stat, float* __restrict__ sums) {
  int b = blockIdx.x * blockDim.x + threadIdx.x;
  if (b >= BB) return;
  float s = 0.0f;
  for (int n = 0; n < NNP; ++n) s += stat[(size_t)b * NNP + n];
  sums[b] = s;
}

__device__ inline float det3c(const float M[3][3]) {
  return M[0][0] * (M[1][1] * M[2][2] - M[1][2] * M[2][1])
       - M[0][1] * (M[1][0] * M[2][2] - M[1][2] * M[2][0])
       + M[0][2] * (M[1][0] * M[2][1] - M[1][1] * M[2][0]);
}

// weighted Kabsch, literal replication of the reference (Jacobi eig of H^T H,
// descending singular values, flip row 2 of V on negative det, R = Vc U^T)
__global__ void kabsch_kernel(const float* __restrict__ p1, const float* __restrict__ flow,
                              const float* __restrict__ stat, const float* __restrict__ sums,
                              float* __restrict__ Rt, float* __restrict__ out_trans) {
  int b = blockIdx.x * blockDim.x + threadIdx.x;
  if (b >= BB) return;
  const float* P = p1   + (size_t)b * NNP * 3;
  const float* F = flow + (size_t)b * NNP * 3;
  const float* S = stat + (size_t)b * NNP;
  float inv = 1.0f / sums[b];
  float cA[3] = {0, 0, 0}, cB[3] = {0, 0, 0};
  for (int n = 0; n < NNP; ++n) {
    float w = S[n] * inv;
    for (int i = 0; i < 3; ++i) {
      float a = P[n * 3 + i];
      cA[i] += w * a;
      cB[i] += w * (a + F[n * 3 + i]);
    }
  }
  float H[3][3] = {{0,0,0},{0,0,0},{0,0,0}};
  for (int n = 0; n < NNP; ++n) {
    float w = S[n] * inv;
    float a[3], bp[3];
    for (int i = 0; i < 3; ++i) {
      a[i]  = P[n * 3 + i] - cA[i];
      bp[i] = P[n * 3 + i] + F[n * 3 + i] - cB[i];
    }
    for (int i = 0; i < 3; ++i)
      for (int j = 0; j < 3; ++j) H[i][j] += w * a[i] * bp[j];
  }
  float A[3][3], V[3][3];
  for (int i = 0; i < 3; ++i)
    for (int j = 0; j < 3; ++j) {
      float s = 0;
      for (int k = 0; k < 3; ++k) s += H[k][i] * H[k][j];
      A[i][j] = s;
      V[i][j] = (i == j) ? 1.0f : 0.0f;
    }
  for (int sweep = 0; sweep < 24; ++sweep) {
    for (int p = 0; p < 2; ++p)
      for (int q = p + 1; q < 3; ++q) {
        float apq = A[p][q];
        if (fabsf(apq) < 1e-24f) continue;
        float tau = (A[q][q] - A[p][p]) / (2.0f * apq);
        float tt  = ((tau >= 0.0f) ? 1.0f : -1.0f) / (fabsf(tau) + sqrtf(1.0f + tau * tau));
        float cc  = 1.0f / sqrtf(1.0f + tt * tt);
        float ss  = tt * cc;
        for (int k = 0; k < 3; ++k) { float u = A[k][p], v = A[k][q]; A[k][p] = cc*u - ss*v; A[k][q] = ss*u + cc*v; }
        for (int k = 0; k < 3; ++k) { float u = A[p][k], v = A[q][k]; A[p][k] = cc*u - ss*v; A[q][k] = ss*u + cc*v; }
        for (int k = 0; k < 3; ++k) { float u = V[k][p], v = V[k][q]; V[k][p] = cc*u - ss*v; V[k][q] = ss*u + cc*v; }
      }
  }
  float w3[3] = {A[0][0], A[1][1], A[2][2]};
  for (int i = 0; i < 2; ++i)
    for (int j = i + 1; j < 3; ++j)
      if (w3[j] > w3[i]) {
        float tw = w3[i]; w3[i] = w3[j]; w3[j] = tw;
        for (int k = 0; k < 3; ++k) { float tv = V[k][i]; V[k][i] = V[k][j]; V[k][j] = tv; }
      }
  float U[3][3], sv[3];
  for (int j = 0; j < 3; ++j) sv[j] = sqrtf(fmaxf(w3[j], 0.0f));
  for (int j = 0; j < 3; ++j) {
    if (j == 2 && sv[2] <= 1e-10f) {
      U[0][2] = U[1][0] * U[2][1] - U[2][0] * U[1][1];
      U[1][2] = U[2][0] * U[0][1] - U[0][0] * U[2][1];
      U[2][2] = U[0][0] * U[1][1] - U[1][0] * U[0][1];
    } else {
      float invs = (sv[j] > 1e-20f) ? 1.0f / sv[j] : 0.0f;
      for (int i = 0; i < 3; ++i) {
        float s = 0;
        for (int k = 0; k < 3; ++k) s += H[i][k] * V[k][j];
        U[i][j] = s * invs;
      }
    }
  }
  float flip = (det3c(U) * det3c(V) < 0.0f) ? -1.0f : 1.0f;
  float Vc[3][3];
  for (int i = 0; i < 3; ++i)
    for (int j = 0; j < 3; ++j) Vc[i][j] = V[i][j] * ((i == 2) ? flip : 1.0f);
  float R[3][3], t3[3];
  for (int i = 0; i < 3; ++i)
    for (int j = 0; j < 3; ++j) {
      float s = 0;
      for (int k = 0; k < 3; ++k) s += Vc[i][k] * U[j][k];
      R[i][j] = s;
    }
  for (int i = 0; i < 3; ++i) {
    float s = 0;
    for (int j = 0; j < 3; ++j) s += R[i][j] * cA[j];
    t3[i] = cB[i] - s;
  }
  for (int i = 0; i < 3; ++i) {
    for (int j = 0; j < 3; ++j) Rt[(size_t)b * 12 + i * 3 + j] = R[i][j];
    Rt[(size_t)b * 12 + 9 + i] = t3[i];
  }
  float* T = out_trans + (size_t)b * 16;
  for (int i = 0; i < 3; ++i) {
    T[i * 4 + 0] = R[i][0]; T[i * 4 + 1] = R[i][1]; T[i * 4 + 2] = R[i][2]; T[i * 4 + 3] = t3[i];
  }
  T[12] = 0.0f; T[13] = 0.0f; T[14] = 0.0f; T[15] = 1.0f;
}

__global__ void finalize_kernel(const float* __restrict__ p1, const float* __restrict__ flow,
                                const float* __restrict__ stat, const float* __restrict__ Rt,
                                float* __restrict__ sf_out, float* __restrict__ stat_out,
                                float* __restrict__ mask_out) {
  int i = blockIdx.x * blockDim.x + threadIdx.x;
  if (i >= BB * NNP) return;
  int b = i / NNP, n = i % NNP;
  const float* R = Rt + (size_t)b * 12;
  const float* tt = R + 9;
  float px = p1[(size_t)i * 3], py = p1[(size_t)i * 3 + 1], pz = p1[(size_t)i * 3 + 2];
  float r0 = R[0] * px + R[1] * py + R[2] * pz + tt[0] - px;
  float r1 = R[3] * px + R[4] * py + R[5] * pz + tt[1] - py;
  float r2 = R[6] * px + R[7] * py + R[8] * pz + tt[2] - pz;
  float sc = stat[i];
  bool mk = sc > 0.5f;
  float s0 = mk ? r0 : flow[(size_t)i * 3];
  float s1 = mk ? r1 : flow[(size_t)i * 3 + 1];
  float s2 = mk ? r2 : flow[(size_t)i * 3 + 2];
  sf_out[((size_t)b * 3 + 0) * NNP + n] = s0;
  sf_out[((size_t)b * 3 + 1) * NNP + n] = s1;
  sf_out[((size_t)b * 3 + 2) * NNP + n] = s2;
  stat_out[(size_t)b * NNP + n] = sc;
  mask_out[i] = mk ? 1.0f : 0.0f;
}

__global__ void copy_kernel(const float* __restrict__ src, float* __restrict__ dst, int n) {
  int i = blockIdx.x * blockDim.x + threadIdx.x;
  if (i < n) dst[i] = src[i];
}

// ---------------------------------------------------------------------------
// Host-side orchestration
// ---------------------------------------------------------------------------
static inline int kpad32(int K) { return (K + 31) & ~31; }

static inline void conv_w(const float* W, _Float16* wt, int K, int N, int Kpad,
                          hipStream_t st) {
  int total = N * Kpad;
  convert_w_kernel<<<dim3((unsigned)((total + 255) / 256)), 256, 0, st>>>(W, wt, K, N, Kpad);
}

static inline dim3 gemm_grid(int M, int Nc) {
  return dim3((unsigned)((M + 15) / 16), (unsigned)((Nc + 63) / 64));
}

static inline void gemm(const float* A, const float* W, const float* bias, float* C,
                        int M, int K, int Nc, int ldC, int act,
                        _Float16* wt, hipStream_t st) {
  int Kp = kpad32(K);
  conv_w(W, wt, K, Nc, Kp, st);
  DenseA al{A, K};
  gemm_wmma<DenseA><<<gemm_grid(M, Nc), 32, 0, st>>>(al, wt, bias, C, M, K, Kp, Nc, ldC, act);
}

extern "C" void kernel_launch(void* const* d_in, const int* in_sizes, int n_in,
                              void* d_out, int out_size, void* d_ws, size_t ws_size,
                              hipStream_t stream) {
  (void)in_sizes; (void)n_in; (void)out_size; (void)ws_size;
  // ---- unflatten inputs (JAX pytree order: dict keys sorted) ----
  int t = 0;
  auto in = [&]() -> const float* { return (const float*)d_in[t++]; };
  const float* feature1 = in();
  const float* feature2 = in();
  const float* gfeat    = in();
  const float *fcW[3], *fcB[3];
  for (int l = 0; l < 3; ++l) { fcW[l] = in(); fcB[l] = in(); }
  const float *fpW[3], *fpB[3];
  for (int l = 0; l < 3; ++l) { fpW[l] = in(); fpB[l] = in(); }
  const float* fpOut = in();
  const float* gru_bhh = in();
  const float* gru_bih = in();
  const float* gru_whh = in();
  const float* gru_wih = in();
  const float *mpW[3], *mpB[3];
  for (int l = 0; l < 3; ++l) { mpW[l] = in(); mpB[l] = in(); }
  const float* mpOut = in();
  const float *m1W[4][3], *m1B[4][3], *m1W2[4][3], *m1B2[4][3];
  for (int s = 0; s < 4; ++s) {
    for (int l = 0; l < 3; ++l) { m1W[s][l]  = in(); m1B[s][l]  = in(); }
    for (int l = 0; l < 3; ++l) { m1W2[s][l] = in(); m1B2[s][l] = in(); }
  }
  const float *m2W[4][3], *m2B[4][3], *m2W2[4][3], *m2B2[4][3];
  for (int s = 0; s < 4; ++s) {
    for (int l = 0; l < 3; ++l) { m2W[s][l]  = in(); m2B[s][l]  = in(); }
    for (int l = 0; l < 3; ++l) { m2W2[s][l] = in(); m2B2[s][l] = in(); }
  }
  const float* pc1 = in();
  const float* pc2 = in();

  // ---- workspace carve (floats, 64B-aligned blocks) ----
  float* wsp = (float*)d_ws;
  size_t off = 0;
  auto alloc = [&](size_t nf) {
    nf = (nf + 15) & ~(size_t)15;
    float* p = wsp + off; off += nf; return p;
  };
  float* p1   = alloc((size_t)BB * NNP * 3);
  float* p2   = alloc((size_t)BB * NNP * 3);
  float* ft1  = alloc((size_t)BB * NNP * 3);
  float* ft2  = alloc((size_t)BB * NNP * 3);
  float* f1   = alloc((size_t)BB * NNP * 512);
  float* f2   = alloc((size_t)BB * NNP * 512);
  float* cor  = alloc((size_t)BB * NNP * 512);
  float* emb  = alloc((size_t)BB * NNP * 1027);
  float* prop = alloc((size_t)BB * NNP * 256);
  float* t1   = alloc((size_t)BB * NNP * 32 * 512);  // biggest transient
  float* t2   = alloc((size_t)BB * NNP * 32 * 256);
  float* t3   = alloc((size_t)BB * NNP * 32 * 64);
  float* pooled = alloc((size_t)BB * NNP * 64);
  float* u1   = alloc((size_t)BB * NNP * 64);
  float* u2   = alloc((size_t)BB * NNP * 64);
  float* finbuf  = alloc((size_t)BB * NNP * 512);
  float* flowbuf = alloc((size_t)BB * NNP * 3);
  float* statbuf = alloc((size_t)BB * NNP);
  float* gmax = alloc(BB * 256);
  float* gi   = alloc(BB * 768);
  float* gh   = alloc(BB * 768);
  float* hbuf = alloc(BB * 256);
  float* sums = alloc(8);
  float* Rt   = alloc(BB * 12 + 8);
  int*   idxbuf = (int*)alloc((size_t)BB * NNP * 32);
  _Float16* wt = (_Float16*)alloc(280000);  // >= 512*1056 f16 = 270336 floats/2

  const float radii[4] = {2.0f, 4.0f, 8.0f, 16.0f};
  const int   nsmp[4]  = {4, 8, 16, 32};
  auto grid1 = [](int n) { return dim3((unsigned)((n + 255) / 256)); };

  // ---- layout transposes ----
  transpose_kernel<<<grid1(BB * 3 * NNP), 256, 0, stream>>>(pc1, p1);
  transpose_kernel<<<grid1(BB * 3 * NNP), 256, 0, stream>>>(pc2, p2);
  transpose_kernel<<<grid1(BB * 3 * NNP), 256, 0, stream>>>(feature1, ft1);
  transpose_kernel<<<grid1(BB * 3 * NNP), 256, 0, stream>>>(feature2, ft2);

  // ---- mse1 on (p, ft) -> f (B,N,512) with max-broadcast upper half ----
  auto run_mse1 = [&](const float* P, const float* FT, float* fout) {
    for (int s = 0; s < 4; ++s) {
      int ns = nsmp[s]; int rows = BB * NNP * ns;
      ball_query_kernel<<<grid1(BB * NNP), 256, 0, stream>>>(P, idxbuf, radii[s] * radii[s], ns);
      GatherPLF g{P, FT, idxbuf, ns, 3};
      int Kp = kpad32(6);
      conv_w(m1W[s][0], wt, 6, 32, Kp, stream);
      gemm_wmma<GatherPLF><<<gemm_grid(rows, 32), 32, 0, stream>>>(
          g, wt, m1B[s][0], t1, rows, 6, Kp, 32, 32, 1);
      gemm(t1, m1W[s][1], m1B[s][1], t2, rows, 32, 32, 32, 1, wt, stream);
      gemm(t2, m1W[s][2], m1B[s][2], t1, rows, 32, 64, 64, 1, wt, stream);
      group_max_kernel<<<grid1(BB * NNP * 64), 256, 0, stream>>>(t1, pooled, BB * NNP, ns, 64);
      gemm(pooled, m1W2[s][0], m1B2[s][0], u1, BB * NNP, 64, 64, 64, 1, wt, stream);
      gemm(u1, m1W2[s][1], m1B2[s][1], u2, BB * NNP, 64, 64, 64, 1, wt, stream);
      gemm(u2, m1W2[s][2], m1B2[s][2], fout + 64 * s, BB * NNP, 64, 64, 512, 1, wt, stream);
    }
    colmax_bcast_kernel<<<grid1(BB * 256), 256, 0, stream>>>(fout, 256);
  };
  run_mse1(p1, ft1, f1);
  run_mse1(p2, ft2, f2);

  // ---- feature correlator (knn-8, fused concat GEMM) ----
  knn_kernel<<<grid1(BB * NNP), 256, 0, stream>>>(p1, p2, idxbuf);
  {
    int rows = BB * NNP * 8;
    GatherCorr g{p1, p2, f1, f2, idxbuf, 8, 512};
    int Kp = kpad32(1027);
    conv_w(fcW[0], wt, 1027, 512, Kp, stream);
    gemm_wmma<GatherCorr><<<gemm_grid(rows, 512), 32, 0, stream>>>(
        g, wt, fcB[0], t1, rows, 1027, Kp, 512, 512, 1);
    gemm(t1, fcW[1], fcB[1], t2, rows, 512, 512, 512, 1, wt, stream);
    gemm(t2, fcW[2], fcB[2], t1, rows, 512, 512, 512, 1, wt, stream);
    group_max_kernel<<<grid1(BB * NNP * 512), 256, 0, stream>>>(t1, cor, BB * NNP, 8, 512);
  }
  build_emb_kernel<<<grid1(BB * NNP * 1027), 256, 0, stream>>>(ft1, f1, cor, emb);

  // ---- mse2 on (p1, emb) -> prop (B,N,256) ----
  for (int s = 0; s < 4; ++s) {
    int ns = nsmp[s]; int rows = BB * NNP * ns;
    ball_query_kernel<<<grid1(BB * NNP), 256, 0, stream>>>(p1, idxbuf, radii[s] * radii[s], ns);
    GatherPLF g{p1, emb, idxbuf, ns, 1027};
    int Kp = kpad32(1030);
    conv_w(m2W[s][0], wt, 1030, 512, Kp, stream);
    gemm_wmma<GatherPLF><<<gemm_grid(rows, 512), 32, 0, stream>>>(
        g, wt, m2B[s][0], t1, rows, 1030, Kp, 512, 512, 1);
    gemm(t1, m2W[s][1], m2B[s][1], t2, rows, 512, 256, 256, 1, wt, stream);
    gemm(t2, m2W[s][2], m2B[s][2], t3, rows, 256, 64, 64, 1, wt, stream);
    group_max_kernel<<<grid1(BB * NNP * 64), 256, 0, stream>>>(t3, pooled, BB * NNP, ns, 64);
    gemm(pooled, m2W2[s][0], m2B2[s][0], u1, BB * NNP, 64, 64, 64, 1, wt, stream);
    gemm(u1, m2W2[s][1], m2B2[s][1], u2, BB * NNP, 64, 64, 64, 1, wt, stream);
    gemm(u2, m2W2[s][2], m2B2[s][2], prop + 64 * s, BB * NNP, 64, 64, 256, 1, wt, stream);
  }

  // ---- GRU ----
  max_over_n_kernel<<<grid1(BB * 256), 256, 0, stream>>>(prop, gmax, 256);
  gemm(gmax,  gru_wih, gru_bih, gi, BB, 256, 768, 768, 0, wt, stream);
  gemm(gfeat, gru_whh, gru_bhh, gh, BB, 256, 768, 768, 0, wt, stream);
  gru_combine_kernel<<<grid1(BB * 256), 256, 0, stream>>>(gi, gh, gfeat, hbuf);

  // ---- heads ----
  build_final_kernel<<<grid1(BB * NNP * 512), 256, 0, stream>>>(prop, hbuf, finbuf);
  gemm(finbuf, fpW[0], fpB[0], t1, BB * NNP, 512, 256, 256, 1, wt, stream);
  gemm(t1, fpW[1], fpB[1], t2, BB * NNP, 256, 128, 128, 1, wt, stream);
  gemm(t2, fpW[2], fpB[2], t3, BB * NNP, 128, 64, 64, 1, wt, stream);
  gemm(t3, fpOut, nullptr, flowbuf, BB * NNP, 64, 3, 3, 0, wt, stream);
  gemm(finbuf, mpW[0], mpB[0], t1, BB * NNP, 512, 256, 256, 1, wt, stream);
  gemm(t1, mpW[1], mpB[1], t2, BB * NNP, 256, 128, 128, 1, wt, stream);
  gemm(t2, mpW[2], mpB[2], t3, BB * NNP, 128, 64, 64, 1, wt, stream);
  gemm(t3, mpOut, nullptr, statbuf, BB * NNP, 64, 1, 1, 2, wt, stream);  // fused sigmoid

  // ---- Kabsch + outputs: [sf(3072) | stat(1024) | trans(32) | mask(1024) | h(512)] ----
  float* out = (float*)d_out;
  score_sum_kernel<<<1, 32, 0, stream>>>(statbuf, sums);
  kabsch_kernel<<<1, 32, 0, stream>>>(p1, flowbuf, statbuf, sums, Rt, out + 4096);
  finalize_kernel<<<grid1(BB * NNP), 256, 0, stream>>>(p1, flowbuf, statbuf, Rt,
                                                       out, out + 3072, out + 4128);
  copy_kernel<<<grid1(BB * 256), 256, 0, stream>>>(hbuf, out + 5152, BB * 256);
}
